// ScaledDotProductAttention_42829413876092
// MI455X (gfx1250) — compile-verified, matrix operand order checked by reading, not run on hardware
//
#include <hip/hip_runtime.h>
#include <hip/hip_bf16.h>
#include <stdint.h>

// ---------------------------------------------------------------------------
// Scaled dot-product attention, MI455X (gfx1250, wave32, WMMA).
//   scores = (Q K^T) * 0.5 ; scores[mask] = -1e9 ; attn = softmax(scores)
//   context = attn @ V ; outputs = (context, attn) concatenated in d_out.
// One wave per 16-row query tile; f16 WMMA (f32 accum) for both GEMMs; full
// score rows kept f16 in the 320KB WGP LDS (single QK^T pass); attn written
// once, coalesced, as f32 (dominant 402MB stream); V tiles staged into LDS
// via gfx1250 async-to-LDS DMA (global_load_async_to_lds_b128, ASYNCcnt).
// ---------------------------------------------------------------------------

typedef __attribute__((ext_vector_type(16))) _Float16 v16h;
typedef __attribute__((ext_vector_type(8)))  float    v8f;
typedef __attribute__((ext_vector_type(4)))  int      v4i;

// async-to-LDS builtin pointer types: global (AS1) src, LDS (AS3) dst
typedef v4i __attribute__((address_space(1)))* gv4i_p;
typedef v4i __attribute__((address_space(3)))* lv4i_p;

#define A_B 2
#define A_H 12
#define A_S 2048
#define A_D 64
#define SCALE 0.5f
#define MASK_FILL -1.0e9f

#define WAVES_PER_BLOCK 4          // 4 waves * 16 rows = 64 query rows / block
#define VSTRIDE_F 68               // padded floats per staged V row (272B, 16B-mult)
#define PROBS_HALVES (WAVES_PER_BLOCK * 16 * A_S)
#define SMEM_BYTES (PROBS_HALVES * 2 + 32 * VSTRIDE_F * 4)

#if defined(__has_builtin)
#if __has_builtin(__builtin_amdgcn_global_load_async_to_lds_b128)
#define HAVE_ASYNC_LDS 1
#endif
#endif

static __device__ __forceinline__ void wait_async_lds() {
#if defined(__has_builtin) && __has_builtin(__builtin_amdgcn_s_wait_asynccnt)
  __builtin_amdgcn_s_wait_asynccnt(0);
#else
  asm volatile("s_wait_asynccnt 0" ::: "memory");
#endif
}

__global__ void
__launch_bounds__(WAVES_PER_BLOCK * 32)
attn_fused_kernel(const float* __restrict__ q,
                  const float* __restrict__ k,
                  const float* __restrict__ v,
                  const uint8_t* __restrict__ mask,   // jnp.bool_ = 1 byte/elem
                  float* __restrict__ ctx_out,        // [B,H,S,D]
                  float* __restrict__ attn_out) {     // [B,H,S,S]
  extern __shared__ char smem[];
  _Float16* probs  = (_Float16*)smem;                       // [4][16][2048] f16
  float*    vstage = (float*)(smem + PROBS_HALVES * 2);     // [32][VSTRIDE_F] f32

  const int wave = threadIdx.x >> 5;
  const int lane = threadIdx.x & 31;
  const int tile = blockIdx.x * WAVES_PER_BLOCK + wave;     // 16-row q tile
  const int bh   = tile >> 7;                               // tile / (S/16)
  const int m0   = (tile & 127) << 4;                       // first q row

  const float*   qb = q + (size_t)bh * A_S * A_D;
  const float*   kb = k + (size_t)bh * A_S * A_D;
  const float*   vb = v + (size_t)bh * A_S * A_D;
  const uint8_t* mb = mask + (size_t)bh * A_S * A_S;
  float* attnb = attn_out + (size_t)bh * A_S * A_S;
  float* ctxb  = ctx_out  + (size_t)bh * A_S * A_D;

  const int mrow = lane & 15;       // row (A) / col (B,C) within 16-tile
  const int hi   = lane >> 4;       // half-wave select
  const int koff = hi * 16;         // K-dim offset held by this lane

  _Float16* myprobs = probs + wave * 16 * A_S;

  // ---- Q fragments: lane holds 16 contiguous head-dims of one Q row ------
  v16h qa[2];
#pragma unroll
  for (int c = 0; c < 2; ++c) {
    const float* p = qb + (size_t)(m0 + mrow) * A_D + c * 32 + koff;
#pragma unroll
    for (int e = 0; e < 16; ++e) qa[c][e] = (_Float16)p[e];
  }

  // ---- Pass 1: scores = scale * Q K^T, masked, stored f16 in LDS ---------
  for (int nt = 0; nt < A_S / 16; ++nt) {
    const int krow = nt * 16 + mrow;
    if (nt + 1 < A_S / 16)      // global_prefetch_b8 on next K tile
      __builtin_prefetch(kb + (size_t)(krow + 16) * A_D + koff, 0, 1);

    v16h kf[2];
#pragma unroll
    for (int c = 0; c < 2; ++c) {
      const float* p = kb + (size_t)krow * A_D + c * 32 + koff;
#pragma unroll
      for (int e = 0; e < 16; ++e) kf[c][e] = (_Float16)p[e];
    }

    v8f acc = {};
    acc = __builtin_amdgcn_wmma_f32_16x16x32_f16(false, qa[0], false, kf[0],
                                                 (short)0, acc, false, false);
    acc = __builtin_amdgcn_wmma_f32_16x16x32_f16(false, qa[1], false, kf[1],
                                                 (short)0, acc, false, false);

    const int col = nt * 16 + mrow;   // C layout: N = lane&15
#pragma unroll
    for (int j = 0; j < 8; ++j) {
      const int r = j + 8 * hi;       // C layout: M = j + 8*(lane>=16)
      float s = acc[j] * SCALE;
      if (mb[(size_t)(m0 + r) * A_S + col]) s = MASK_FILL;
      myprobs[r * A_S + col] = (_Float16)s;
    }
  }

  // ---- Softmax per row: max -> exp+sum (exp stored back) -> normalize ----
  for (int r = 0; r < 16; ++r) {
    _Float16* row = myprobs + r * A_S;

    float mx = -3.4e38f;
    for (int c0 = lane * 4; c0 < A_S; c0 += 128) {     // ds_load_b64 sweeps
#pragma unroll
      for (int e = 0; e < 4; ++e) mx = fmaxf(mx, (float)row[c0 + e]);
    }
#pragma unroll
    for (int off = 16; off > 0; off >>= 1)
      mx = fmaxf(mx, __shfl_xor(mx, off, 32));

    float sum = 0.0f;
    for (int c0 = lane * 4; c0 < A_S; c0 += 128) {
#pragma unroll
      for (int e = 0; e < 4; ++e) {
        const float ex = __expf((float)row[c0 + e] - mx);
        sum += ex;
        row[c0 + e] = (_Float16)ex;   // keep unnormalized exp in LDS
      }
    }
#pragma unroll
    for (int off = 16; off > 0; off >>= 1)
      sum += __shfl_xor(sum, off, 32);
    const float inv = 1.0f / sum;

    float* orow = attnb + (size_t)(m0 + r) * A_S;
    for (int c0 = lane * 4; c0 < A_S; c0 += 128) {
      float4 pr;
      pr.x = (float)row[c0 + 0] * inv;
      pr.y = (float)row[c0 + 1] * inv;
      pr.z = (float)row[c0 + 2] * inv;
      pr.w = (float)row[c0 + 3] * inv;
      *(float4*)(orow + c0) = pr;     // coalesced 16B attn stores
      row[c0 + 0] = (_Float16)pr.x;   // f16 probs for the PV GEMM
      row[c0 + 1] = (_Float16)pr.y;
      row[c0 + 2] = (_Float16)pr.z;
      row[c0 + 3] = (_Float16)pr.w;
    }
  }

  __syncthreads();

  // ---- Pass 2: context = P @ V, V tile staged in LDS (async DMA) ---------
  v8f cacc[4] = {};                   // 16x64 f32 accumulator (4 x 16x16)
  for (int kt = 0; kt < A_S / 32; ++kt) {
    // stage V[kt*32 .. kt*32+31][0..63] as f32 into LDS
#if defined(HAVE_ASYNC_LDS)
    for (int ch = threadIdx.x; ch < 512; ch += WAVES_PER_BLOCK * 32) {
      const int row = ch >> 4, c4 = (ch & 15) * 4;        // 16B chunks
      const float* gp = vb + (size_t)(kt * 32 + row) * A_D + c4;
      float*       lp = vstage + row * VSTRIDE_F + c4;
      __builtin_amdgcn_global_load_async_to_lds_b128(
          (gv4i_p)gp, (lv4i_p)lp, 0, 0);
    }
    wait_async_lds();
#else
    for (int i = threadIdx.x; i < 32 * A_D; i += WAVES_PER_BLOCK * 32) {
      const int kk = i >> 6, dd = i & 63;
      vstage[kk * VSTRIDE_F + dd] = vb[(size_t)(kt * 32 + kk) * A_D + dd];
    }
#endif
    __syncthreads();

    // A fragment: 16 contiguous prob halves of this lane's row
    const v16h pa = *(const v16h*)(myprobs + mrow * A_S + kt * 32 + koff);

#pragma unroll
    for (int c = 0; c < 4; ++c) {     // 4 chunks of 16 output dims
      v16h vf;
#pragma unroll
      for (int j = 0; j < 16; ++j)    // B layout: lane = col N, elem j = K
        vf[j] = (_Float16)vstage[(koff + j) * VSTRIDE_F + c * 16 + mrow];
      cacc[c] = __builtin_amdgcn_wmma_f32_16x16x32_f16(
          false, pa, false, vf, (short)0, cacc[c], false, false);
    }
    __syncthreads();
  }

  // ---- write context ------------------------------------------------------
#pragma unroll
  for (int c = 0; c < 4; ++c)
#pragma unroll
    for (int j = 0; j < 8; ++j)
      ctxb[(size_t)(m0 + j + 8 * hi) * A_D + c * 16 + mrow] = cacc[c][j];
}

extern "C" void kernel_launch(void* const* d_in, const int* in_sizes, int n_in,
                              void* d_out, int out_size, void* d_ws, size_t ws_size,
                              hipStream_t stream) {
  (void)in_sizes; (void)n_in; (void)d_ws; (void)ws_size; (void)out_size;
  const float*   q    = (const float*)d_in[0];
  const float*   k    = (const float*)d_in[1];
  const float*   v    = (const float*)d_in[2];
  const uint8_t* mask = (const uint8_t*)d_in[3];   // jnp.bool_ buffer

  float* ctx  = (float*)d_out;                               // [B,H,S,D]
  float* attn = ctx + (size_t)A_B * A_H * A_S * A_D;         // [B,H,S,S]

  const int total_tiles = A_B * A_H * (A_S / 16);            // 3072
  dim3 grid(total_tiles / WAVES_PER_BLOCK);                  // 768 blocks
  dim3 block(WAVES_PER_BLOCK * 32);                          // 128 thr = 4 waves
  attn_fused_kernel<<<grid, block, SMEM_BYTES, stream>>>(q, k, v, mask,
                                                         ctx, attn);
}
